// BiLSTMLayer_14345190768765
// MI455X (gfx1250) — matrix operand order repaired and checked by reading, not compiled, not run on hardware
//
#include <hip/hip_runtime.h>
#include <hip/hip_bf16.h>
#include <math.h>

// BiLSTM for MI455X (gfx1250, wave32, WMMA).
// Phase 1: x_proj = x @ Wih^T + bih (both dirs), WMMA f16 GEMM, f32 accum,
//          stored f32 in d_ws in TRANSPOSED layout xpT[t][gcol][b] so the
//          sequential phase reads it with contiguous b128 loads.
// Phase 2: one persistent 512-thread workgroup per direction. Whh lives in
//          registers as f16 WMMA B-fragments (256 VGPR/wave), h in LDS (f16),
//          gates exchanged via a transposed LDS buffer; 2048 sequential steps,
//          32 v_wmma per wave per step.

#define B_DIM 8
#define S_LEN 2048
#define D_IN  256
#define H_DIM 256
#define G4    1024            // 4*H
#define M_TOT (B_DIM * S_LEN) // 16384
#define XPT_T (G4 * B_DIM)    // 8192 floats per timestep in xpT

typedef _Float16 v16h __attribute__((ext_vector_type(16)));
typedef _Float16 h8   __attribute__((ext_vector_type(8)));
typedef float    v8f  __attribute__((ext_vector_type(8)));
typedef float    f4   __attribute__((ext_vector_type(4)));

static __device__ __forceinline__ v8f zero8() {
    v8f z;
#pragma unroll
    for (int i = 0; i < 8; ++i) z[i] = 0.0f;
    return z;
}

// A fragment (16x32 f16) from row-major f32 source, ld = row stride.
// ISA layout: lane m = lane&15; lanes<16 hold K={0..7}U{16..23},
// lanes>=16 hold K={8..15}U{24..31}; element e<8 -> run0, e>=8 -> run1.
static __device__ __forceinline__ v16h frag_a_global(const float* __restrict__ X,
                                                     int row0, int ld, int k0, int lane) {
    int m  = lane & 15;
    int hi = lane >> 4;
    int r0 = hi ? 8 : 0;
    int r1 = hi ? 24 : 16;
    const float* p = X + (size_t)(row0 + m) * ld + k0;
    f4 a0 = *(const f4*)(p + r0);
    f4 a1 = *(const f4*)(p + r0 + 4);
    f4 b0 = *(const f4*)(p + r1);
    f4 b1 = *(const f4*)(p + r1 + 4);
    v16h v;
#pragma unroll
    for (int i = 0; i < 4; ++i) {
        v[i]      = (_Float16)a0[i];
        v[4 + i]  = (_Float16)a1[i];
        v[8 + i]  = (_Float16)b0[i];
        v[12 + i] = (_Float16)b1[i];
    }
    return v;
}

// A fragment (16x32 f16) from LDS f16 h-buffer, ld = 256 halves.
static __device__ __forceinline__ v16h frag_a_lds(const _Float16* Hb, int k0, int lane) {
    int m  = lane & 15;
    int hi = lane >> 4;
    int r0 = hi ? 8 : 0;
    int r1 = hi ? 24 : 16;
    h8 lo = *(const h8*)(Hb + m * H_DIM + k0 + r0);
    h8 hv = *(const h8*)(Hb + m * H_DIM + k0 + r1);
    v16h v;
#pragma unroll
    for (int i = 0; i < 8; ++i) { v[i] = lo[i]; v[8 + i] = hv[i]; }
    return v;
}

// B fragment (32x16 f16): B[k,n] = W[n,k] (W row-major [1024 x 256]).
// ISA layout: lane n = lane&15; lanes<16 hold K={0..7}U{16..23}, lanes>=16 K={8..15}U{24..31}.
static __device__ __forceinline__ v16h frag_b_global(const float* __restrict__ W,
                                                     int col0, int ld, int k0, int lane) {
    int n     = lane & 15;
    int kbase = (lane & 16) ? 8 : 0;
    const float* p = W + (size_t)(col0 + n) * ld + k0 + kbase;
    f4 q0 = *(const f4*)(p);
    f4 q1 = *(const f4*)(p + 4);
    f4 q2 = *(const f4*)(p + 16);
    f4 q3 = *(const f4*)(p + 20);
    v16h v;
#pragma unroll
    for (int i = 0; i < 4; ++i) {
        v[i]      = (_Float16)q0[i];
        v[4 + i]  = (_Float16)q1[i];
        v[8 + i]  = (_Float16)q2[i];
        v[12 + i] = (_Float16)q3[i];
    }
    return v;
}

static __device__ __forceinline__ float sigmoidf_dev(float x) {
    return 1.0f / (1.0f + __expf(-x));
}

// ---------------- Phase 1: input projection GEMM -----------------
// grid = (16384/128, 1024/64, 2), block = 256 (8 waves).
// Wave w computes rows [blockM + 16w, +16), cols [blockN, +64), K = 256.
// Epilogue writes the transposed layout xpT[t*8192 + gcol*8 + b].
__global__ __launch_bounds__(256) void bilstm_proj_kernel(
    const float* __restrict__ x,
    const float* __restrict__ Wf, const float* __restrict__ bf,
    const float* __restrict__ Wb, const float* __restrict__ bb,
    float* __restrict__ xpf, float* __restrict__ xpb) {
    const float* W   = blockIdx.z ? Wb : Wf;
    const float* bi  = blockIdx.z ? bb : bf;
    float*       xpT = blockIdx.z ? xpb : xpf;

    int lane = threadIdx.x & 31;
    int w    = threadIdx.x >> 5;               // 0..7
    int row0 = blockIdx.x * 128 + w * 16;      // row = b*2048 + t
    int col0 = blockIdx.y * 64;

    v8f acc[4];
#pragma unroll
    for (int nt = 0; nt < 4; ++nt) acc[nt] = zero8();

#pragma unroll
    for (int kt = 0; kt < 8; ++kt) {
        v16h a = frag_a_global(x, row0, D_IN, kt * 32, lane);
#pragma unroll
        for (int nt = 0; nt < 4; ++nt) {
            v16h b = frag_b_global(W, col0 + nt * 16, D_IN, kt * 32, lane);
            acc[nt] = __builtin_amdgcn_wmma_f32_16x16x32_f16(
                false, a, false, b, (short)0, acc[nt], false, false);
        }
    }

    int n_lane = lane & 15;
    int mofs   = (lane & 16) ? 8 : 0;
#pragma unroll
    for (int nt = 0; nt < 4; ++nt) {
        int   gcol = col0 + nt * 16 + n_lane;
        float bias = bi[gcol];
#pragma unroll
        for (int r = 0; r < 8; ++r) {
            int grow = row0 + r + mofs;        // = b*2048 + t
            int b    = grow >> 11;
            int tt   = grow & (S_LEN - 1);
            xpT[(size_t)tt * XPT_T + gcol * B_DIM + b] = acc[nt][r] + bias;
        }
    }
}

// ---------------- Phase 2: sequential recurrence -----------------
// grid = 2 (one block per direction), block = 512 (16 waves).
// Wave w owns gate columns [64w, 64w+64): 4 N-tiles x 8 K-tiles of Whh as
// persistent f16 WMMA B-fragments (256 VGPRs/wave).
__global__ __launch_bounds__(512) void bilstm_rec_kernel(
    const float* __restrict__ xpf, const float* __restrict__ Whf,
    const float* __restrict__ xpb, const float* __restrict__ Whb,
    float* __restrict__ out) {
    __shared__ alignas(16) _Float16 hbuf[16 * H_DIM];   // h (f16); rows 8..15 stay 0
    __shared__ alignas(16) float    gbufT[G4 * B_DIM];  // activated gates [gcol][b], 32KB

    const int    dir = blockIdx.x;                      // 0 = fwd, 1 = bwd
    const float* xpT = dir ? xpb : xpf;
    const float* Wh  = dir ? Whb : Whf;
    const int    ofs = dir ? H_DIM : 0;

    const int  tid    = threadIdx.x;
    const int  lane   = tid & 31;
    const int  w      = tid >> 5;                       // 0..15
    const int  n_lane = lane & 15;
    const bool lolane = (lane < 16);

    for (int i = tid; i < 16 * H_DIM; i += 512) hbuf[i] = (_Float16)0.0f;

    // persistent Whh B-fragments
    v16h Bf[4][8];
#pragma unroll
    for (int nt = 0; nt < 4; ++nt)
#pragma unroll
        for (int kt = 0; kt < 8; ++kt)
            Bf[nt][kt] = frag_b_global(Wh, w * 64 + nt * 16, H_DIM, kt * 32, lane);

    // cell state: column-owner threads (tid<256) hold c for all 8 batches of col j
    const int j = tid;                                  // hidden column (tid<256)
    f4 clo, chi;
#pragma unroll
    for (int i = 0; i < 4; ++i) { clo[i] = 0.0f; chi[i] = 0.0f; }

    __syncthreads();

    for (int s = 0; s < S_LEN; ++s) {
        const int t = dir ? (S_LEN - 1 - s) : s;

        // x_proj for this step: 2 contiguous b128 loads per N-tile
        // (lanes>=16 mirror lanes<16; values unused for padded rows)
        f4 xlo[4], xhi[4];
#pragma unroll
        for (int nt = 0; nt < 4; ++nt) {
            int gcol = w * 64 + nt * 16 + n_lane;
            const float* p = xpT + (size_t)t * XPT_T + gcol * B_DIM;
            xlo[nt] = *(const f4*)(p);
            xhi[nt] = *(const f4*)(p + 4);
        }
        // prefetch next step's xpT slice (gfx1250 global_prefetch_b8)
        if (s + 1 < S_LEN) {
            int tn = dir ? (t - 1) : (t + 1);
            const float* pf = xpT + (size_t)tn * XPT_T + (w * 64 + lane) * B_DIM;
            __builtin_prefetch(pf, 0, 1);
        }

        // gates_mm = h @ Whh^T
        v8f acc[4];
#pragma unroll
        for (int nt = 0; nt < 4; ++nt) acc[nt] = zero8();
#pragma unroll
        for (int kt = 0; kt < 8; ++kt) {
            v16h a = frag_a_lds(hbuf, kt * 32, lane);
#pragma unroll
            for (int nt = 0; nt < 4; ++nt)
                acc[nt] = __builtin_amdgcn_wmma_f32_16x16x32_f16(
                    false, a, false, Bf[nt][kt], (short)0, acc[nt], false, false);
        }

        // add x_proj, activate (i,f,o: sigmoid; g: tanh), publish transposed:
        // gbufT[gcol*8 + b] via two b128 stores per N-tile (lanes<16 only)
        if (lolane) {
#pragma unroll
            for (int nt = 0; nt < 4; ++nt) {
                int  gcol = w * 64 + nt * 16 + n_lane;
                bool is_g = ((gcol >> 8) == 2);
                f4 v0, v1;
#pragma unroll
                for (int r = 0; r < 4; ++r) {
                    float a0 = acc[nt][r] + xlo[nt][r];
                    float a1 = acc[nt][r + 4] + xhi[nt][r];
                    v0[r] = is_g ? tanhf(a0) : sigmoidf_dev(a0);
                    v1[r] = is_g ? tanhf(a1) : sigmoidf_dev(a1);
                }
                float* g = &gbufT[gcol * B_DIM];
                *(f4*)(g)     = v0;
                *(f4*)(g + 4) = v1;
            }
        }
        __syncthreads();

        // elementwise: thread j (< 256) owns hidden column j for all 8 batches
        if (tid < H_DIM) {
            const float* gi = &gbufT[j * B_DIM];
            const float* gf = &gbufT[(256 + j) * B_DIM];
            const float* gg = &gbufT[(512 + j) * B_DIM];
            const float* go = &gbufT[(768 + j) * B_DIM];
            f4 ilo = *(const f4*)(gi), ihi = *(const f4*)(gi + 4);
            f4 flo = *(const f4*)(gf), fhi = *(const f4*)(gf + 4);
            f4 glo = *(const f4*)(gg), ghi = *(const f4*)(gg + 4);
            f4 olo = *(const f4*)(go), ohi = *(const f4*)(go + 4);
            float hval[8];
#pragma unroll
            for (int i = 0; i < 4; ++i) {
                float cn = flo[i] * clo[i] + ilo[i] * glo[i];
                clo[i]   = cn;
                hval[i]  = olo[i] * tanhf(cn);
            }
#pragma unroll
            for (int i = 0; i < 4; ++i) {
                float cn    = fhi[i] * chi[i] + ihi[i] * ghi[i];
                chi[i]      = cn;
                hval[4 + i] = ohi[i] * tanhf(cn);
            }
#pragma unroll
            for (int m = 0; m < 8; ++m) {
                hbuf[m * H_DIM + j] = (_Float16)hval[m];
                out[(size_t)(m * S_LEN + t) * (2 * H_DIM) + ofs + j] = hval[m];
            }
        }
        __syncthreads();
    }
}

extern "C" void kernel_launch(void* const* d_in, const int* in_sizes, int n_in,
                              void* d_out, int out_size, void* d_ws, size_t ws_size,
                              hipStream_t stream) {
    (void)in_sizes; (void)n_in; (void)out_size; (void)ws_size;
    const float* x     = (const float*)d_in[0];
    const float* Wih_f = (const float*)d_in[1];
    const float* bih_f = (const float*)d_in[2];
    const float* Whh_f = (const float*)d_in[3];
    const float* Wih_b = (const float*)d_in[4];
    const float* bih_b = (const float*)d_in[5];
    const float* Whh_b = (const float*)d_in[6];
    float* out  = (float*)d_out;
    float* xp_f = (float*)d_ws;                       // xpT fwd: [2048][1024][8] f32
    float* xp_b = xp_f + (size_t)S_LEN * XPT_T;       // xpT bwd

    dim3 gp(M_TOT / 128, G4 / 64, 2);                 // 128 x 16 x 2 blocks
    bilstm_proj_kernel<<<gp, 256, 0, stream>>>(x, Wih_f, bih_f, Wih_b, bih_b,
                                               xp_f, xp_b);
    bilstm_rec_kernel<<<2, 512, 0, stream>>>(xp_f, Whh_f, xp_b, Whh_b, out);
}
